// SkillModelVectorQuantizedPriorDist_326417514851
// MI455X (gfx1250) — matrix operand order repaired and checked by reading, not compiled, not run on hardware
//
#include <hip/hip_runtime.h>
#include <math.h>

#define B_ 8192
#define T_ 64
#define S_ 47
#define A_ 8
#define Z_ 16
#define H_ 128
#define K_ 256

// output layout (flat, return order)
#define MEAN_OFF   0
#define SIG_OFF    ((size_t)B_ * S_)                    // 385024
#define AMEAN_OFF  ((size_t)2 * B_ * S_)                // 770048
#define ASIG_OFF   (AMEAN_OFF + (size_t)B_ * T_ * 8)    // 4964352
#define VQ_OFF     (ASIG_OFF + (size_t)B_ * T_ * 8)     // 9158656
#define PRIOR_OFF  (VQ_OFF + 1)                         // 9158657

typedef __attribute__((ext_vector_type(16))) __bf16 v16bf;
typedef __attribute__((ext_vector_type(8)))  float  v8f;

union Frag { v16bf v; unsigned int u[8]; };

struct P { const float* a[39]; };

__device__ __forceinline__ float softplusf(float x) {
    return x > 20.f ? x : log1pf(expf(x));
}

// Stage weight matrix (Krows x Nact, row-major f32) -> LDS bf16 [Npad][Kd] (n-major),
// zero-padded in K and N. Each thread handles 4 consecutive k of one n and emits a
// single aligned 8-byte LDS store -> conflict-free across the wave.
__device__ __forceinline__ void stage_w(const float* __restrict__ gw, int Krows, int Kd,
                                        int Nact, int Npad, __bf16* wb, int tid) {
    const int kc = Kd >> 2;
    const int total = kc * Npad;
    for (int i = tid; i < total; i += 256) {
        int n  = i / kc;
        int k0 = (i - n * kc) << 2;
        union { __bf16 h[4]; unsigned long long q; } tmp;
#pragma unroll
        for (int j = 0; j < 4; ++j) {
            int k = k0 + j;
            float v = (k < Krows && n < Nact) ? gw[k * Nact + n] : 0.f;
            tmp.h[j] = (__bf16)v;
        }
        *(unsigned long long*)(wb + n * Kd + k0) = tmp.q;
    }
}

// C = relu(A(64xKd) @ W(Kdx128) + bias) -> bf16 LDS [64][128]
// A in LDS bf16 [64][sA], W in LDS bf16 n-major [128][Kd].
// 4 accumulators per wave: A fragment loaded once per K-step, reused for 4 N-tiles.
__device__ __forceinline__ void gemm128(const __bf16* __restrict__ aL, int sA, int Kd,
                                        const __bf16* __restrict__ wL,
                                        const float* __restrict__ bias,
                                        __bf16* __restrict__ oL,
                                        int w, int lm, int g) {
    const int mt  = w >> 1;           // 4 M-tiles across 8 waves
    const int nt0 = (w & 1) * 4;      // each wave owns 4 N-tiles
    const int row = mt * 16 + lm;
    v8f c[4];
#pragma unroll
    for (int j = 0; j < 4; ++j) c[j] = (v8f){0.f,0.f,0.f,0.f,0.f,0.f,0.f,0.f};
    for (int ks = 0; ks < Kd; ks += 32) {
        Frag a;
#pragma unroll
        for (int v = 0; v < 8; ++v) {
            // A in contiguous k order (this code's choice, not the ISA table order): lanes 0-15 hold K 0..15, 16-31 K 16..31
            int kbA = ks + 2 * v + 16 * g;
            a.u[v] = *(const unsigned int*)(aL + row * sA + kbA);
        }
#pragma unroll
        for (int j = 0; j < 4; ++j) {
            const int col = (nt0 + j) * 16 + lm;
            Frag bb;
#pragma unroll
            for (int v = 0; v < 8; ++v) {
                // B in the same contiguous k order as A (not the ISA table order): lanes 0-15 hold K 0..15, 16-31 K 16..31
                int kbB = ks + 2 * v + 16 * g;
                bb.u[v] = *(const unsigned int*)(wL + col * Kd + kbB);
            }
            c[j] = __builtin_amdgcn_wmma_f32_16x16x32_bf16(false, a.v, false, bb.v,
                                                           (short)0, c[j], false, false);
        }
    }
#pragma unroll
    for (int j = 0; j < 4; ++j) {
        const int col = (nt0 + j) * 16 + lm;
        float bs = bias[col];
#pragma unroll
        for (int v = 0; v < 8; ++v) {
            float val = c[j][v] + bs;
            val = val > 0.f ? val : 0.f;
            oL[(mt * 16 + v + 8 * g) * H_ + col] = (__bf16)val;
        }
    }
}

// out(64x8, f32 global) = act(A(64x128) @ W(128x8) + bias); W padded to N=16 in LDS.
__device__ __forceinline__ void gemm_out8(const __bf16* __restrict__ aL,
                                          const __bf16* __restrict__ wL,
                                          const float* __restrict__ bias,
                                          float* __restrict__ gout,
                                          int w, int lm, int g, int doSoftplus) {
    if (w >= 4) return;               // 4 M-tiles, 1 N-tile
    const int mt  = w;
    const int row = mt * 16 + lm;
    v8f c = {0.f, 0.f, 0.f, 0.f, 0.f, 0.f, 0.f, 0.f};
    for (int ks = 0; ks < H_; ks += 32) {
        Frag a, bb;
#pragma unroll
        for (int v = 0; v < 8; ++v) {
            int kbA = ks + 2 * v + 16 * g;
            int kbB = ks + 2 * v + 16 * g;
            a.u[v]  = *(const unsigned int*)(aL + row * H_ + kbA);
            bb.u[v] = *(const unsigned int*)(wL + lm * H_ + kbB);
        }
        c = __builtin_amdgcn_wmma_f32_16x16x32_bf16(false, a.v, false, bb.v,
                                                    (short)0, c, false, false);
    }
    if (lm < 8) {
        float bs = bias[lm];
#pragma unroll
        for (int v = 0; v < 8; ++v) {
            int r = mt * 16 + v + 8 * g;
            float val = c[v] + bs;
            if (doSoftplus) val = softplusf(val);
            gout[r * 8 + lm] = val;
        }
    }
}

__global__ __launch_bounds__(256) void skill_fwd(P p, float* __restrict__ out,
                                                 float* __restrict__ vqws) {
    __shared__ __bf16 wbuf[H_ * H_];   // 32 KB, also aliased as f32 scratch between phases
    __shared__ __bf16 actA[T_ * H_];   // 16 KB
    __shared__ __bf16 actB[T_ * H_];   // 16 KB
    float* fs = (float*)wbuf;

    const int tid  = threadIdx.x;
    const int b    = blockIdx.x;
    const int w    = tid >> 5;
    const int lane = tid & 31;
    const int g    = lane >> 4;
    const int lm   = lane & 15;

    const float* states  = p.a[0];
    const float* actions = p.a[1];

    // ---------------- Phase 1: encoder over T=64 tokens ----------------
    for (int i = tid; i < T_ * 64; i += 256) {
        int r = i >> 6, cc = i & 63;
        float v = 0.f;
        if (cc < S_)            v = states[((size_t)b * T_ + r) * S_ + cc];
        else if (cc < S_ + A_)  v = actions[((size_t)b * T_ + r) * A_ + (cc - S_)];
        actB[r * 64 + cc] = (__bf16)v;
    }
    stage_w(p.a[2], S_ + A_, 64, H_, H_, wbuf, tid);           // enc_w1 (55x128 -> 64-pad)
    __syncthreads();
    gemm128(actB, 64, 64, wbuf, p.a[3], actA, w, lm, g);       // h1
    __syncthreads();
    stage_w(p.a[4], H_, H_, H_, H_, wbuf, tid);                // enc_w2
    __syncthreads();
    gemm128(actA, H_, H_, wbuf, p.a[5], actB, w, lm, g);       // h2
    __syncthreads();
    // hbar = mean over T (deterministic sequential column sum)
    if (tid < H_) {
        float s = 0.f;
        for (int r = 0; r < T_; ++r) s += (float)actB[r * H_ + tid];
        fs[tid] = s * (1.f / 64.f);                            // hbar @ fs[0..127]
    }
    __syncthreads();

    float* hbar = fs;
    float* ze   = fs + 128;   // 16
    float* zq   = fs + 144;   // 16
    float* dst  = fs + 160;   // 256
    float* s0   = fs + 416;   // 47
    float* in63 = fs + 464;   // 63
    float* vA   = fs + 528;   // 128
    float* vB   = fs + 656;   // 128
    int*   sidx = (int*)(fs + 784);

    // ---------------- Phase 2: VQ ----------------
    if (tid < Z_) {
        const float* wz = p.a[6];                              // (128,16)
        float s = p.a[7][tid];
        for (int k = 0; k < H_; ++k) s += hbar[k] * wz[k * Z_ + tid];
        ze[tid] = s;
    }
    __syncthreads();
    const float* cb = p.a[8];                                  // (256,16)
    {
        float d = 0.f;
        for (int j = 0; j < Z_; ++j) { float t = ze[j] - cb[tid * Z_ + j]; d += t * t; }
        dst[tid] = d;
    }
    __syncthreads();
    if (tid == 0) {
        int best = 0; float bd = dst[0];
        for (int k = 1; k < K_; ++k) if (dst[k] < bd) { bd = dst[k]; best = k; }
        *sidx = best;
    }
    __syncthreads();
    if (tid < Z_) zq[tid] = cb[(*sidx) * Z_ + tid];
    __syncthreads();
    if (tid == 0) {
        float s = 0.f;
        for (int j = 0; j < Z_; ++j) { float t = ze[j] - zq[j]; s += t * t; }
        vqws[b] = 1.25f * s / (float)((size_t)B_ * Z_);        // (1+BETA)*mean contribution
    }

    // ---------------- Phase 3: prior + abstract dynamics (M=1 matvecs) -------------
    if (tid < S_) s0[tid] = states[(size_t)b * T_ * S_ + tid];
    __syncthreads();
    if (tid < H_) {                                            // prior layer 1
        const float* w1 = p.a[9]; float s = p.a[10][tid];
        for (int k = 0; k < S_; ++k) s += s0[k] * w1[k * H_ + tid];
        vA[tid] = s > 0.f ? s : 0.f;
    }
    __syncthreads();
    if (tid < H_) {                                            // prior layer 2
        const float* w2 = p.a[11]; float s = p.a[12][tid];
        for (int k = 0; k < H_; ++k) s += vA[k] * w2[k * H_ + tid];
        vB[tid] = s > 0.f ? s : 0.f;
    }
    __syncthreads();
    {                                                          // prior logits (256)
        const float* w3 = p.a[13]; float s = p.a[14][tid];
        for (int k = 0; k < H_; ++k) s += vB[k] * w3[k * K_ + tid];
        out[PRIOR_OFF + (size_t)b * K_ + tid] = s;
    }
    if (tid < S_ + Z_) in63[tid] = tid < S_ ? s0[tid] : zq[tid - S_];
    __syncthreads();
    if (tid < H_) {                                            // ad f1
        const float* w1 = p.a[15]; float s = p.a[16][tid];
        for (int k = 0; k < S_ + Z_; ++k) s += in63[k] * w1[k * H_ + tid];
        vA[tid] = s > 0.f ? s : 0.f;
    }
    __syncthreads();
    if (tid < H_) {                                            // ad f2
        const float* w2 = p.a[17]; float s = p.a[18][tid];
        for (int k = 0; k < H_; ++k) s += vA[k] * w2[k * H_ + tid];
        vB[tid] = s > 0.f ? s : 0.f;
    }
    __syncthreads();
    if (tid < H_) {                                            // ad m1
        const float* wm = p.a[19]; float s = p.a[20][tid];
        for (int k = 0; k < H_; ++k) s += vB[k] * wm[k * H_ + tid];
        vA[tid] = s > 0.f ? s : 0.f;
    }
    __syncthreads();
    if (tid < S_) {                                            // sT_mean
        const float* wm = p.a[21]; float s = p.a[22][tid];
        for (int k = 0; k < H_; ++k) s += vA[k] * wm[k * S_ + tid];
        out[MEAN_OFF + (size_t)b * S_ + tid] = s;
    }
    __syncthreads();
    if (tid < H_) {                                            // ad s1
        const float* ws_ = p.a[23]; float s = p.a[24][tid];
        for (int k = 0; k < H_; ++k) s += vB[k] * ws_[k * H_ + tid];
        vA[tid] = s > 0.f ? s : 0.f;
    }
    __syncthreads();
    if (tid < S_) {                                            // sT_sig
        const float* ws_ = p.a[25]; float s = p.a[26][tid];
        for (int k = 0; k < H_; ++k) s += vA[k] * ws_[k * S_ + tid];
        out[SIG_OFF + (size_t)b * S_ + tid] = softplusf(s);
    }
    __syncthreads();

    // ---------------- Phase 4: low-level policy over T=64 tokens ----------------
    // input [64][64]: cols 0..46 states, 47..62 z (=z_q forward), 63 = 0; reads zq
    // BEFORE the next stage_w overwrites the scratch region.
    for (int i = tid; i < T_ * 64; i += 256) {
        int r = i >> 6, cc = i & 63;
        float v = 0.f;
        if (cc < S_)            v = states[((size_t)b * T_ + r) * S_ + cc];
        else if (cc < S_ + Z_)  v = zq[cc - S_];
        actB[r * 64 + cc] = (__bf16)v;
    }
    __syncthreads();
    stage_w(p.a[27], S_ + Z_, 64, H_, H_, wbuf, tid);          // ll_w1
    __syncthreads();
    gemm128(actB, 64, 64, wbuf, p.a[28], actA, w, lm, g);      // g1
    __syncthreads();
    stage_w(p.a[29], H_, H_, H_, H_, wbuf, tid);               // ll_w2
    __syncthreads();
    gemm128(actA, H_, H_, wbuf, p.a[30], actB, w, lm, g);      // g2 (kept in actB)
    __syncthreads();
    stage_w(p.a[31], H_, H_, H_, H_, wbuf, tid);               // ll_mw1
    __syncthreads();
    gemm128(actB, H_, H_, wbuf, p.a[32], actA, w, lm, g);      // m1
    __syncthreads();
    stage_w(p.a[33], H_, H_, A_, 16, wbuf, tid);               // ll_mw2 (N pad 8->16)
    __syncthreads();
    gemm_out8(actA, wbuf, p.a[34], out + AMEAN_OFF + (size_t)b * T_ * 8, w, lm, g, 0);
    __syncthreads();
    stage_w(p.a[35], H_, H_, H_, H_, wbuf, tid);               // ll_sw1
    __syncthreads();
    gemm128(actB, H_, H_, wbuf, p.a[36], actA, w, lm, g);      // sg1
    __syncthreads();
    stage_w(p.a[37], H_, H_, A_, 16, wbuf, tid);               // ll_sw2
    __syncthreads();
    gemm_out8(actA, wbuf, p.a[38], out + ASIG_OFF + (size_t)b * T_ * 8, w, lm, g, 1);
}

// Deterministic tree reduction of per-batch vq contributions.
__global__ __launch_bounds__(256) void vq_reduce(const float* __restrict__ vqws,
                                                 float* __restrict__ out) {
    __shared__ float sm[256];
    float s = 0.f;
    for (int i = threadIdx.x; i < B_; i += 256) s += vqws[i];
    sm[threadIdx.x] = s;
    __syncthreads();
    for (int off = 128; off > 0; off >>= 1) {
        if ((int)threadIdx.x < off) sm[threadIdx.x] += sm[threadIdx.x + off];
        __syncthreads();
    }
    if (threadIdx.x == 0) out[0] = sm[0];
}

extern "C" void kernel_launch(void* const* d_in, const int* in_sizes, int n_in,
                              void* d_out, int out_size, void* d_ws, size_t ws_size,
                              hipStream_t stream) {
    (void)in_sizes; (void)n_in; (void)out_size; (void)ws_size;
    P p;
    for (int i = 0; i < 39; ++i) p.a[i] = (const float*)d_in[i];
    float* out  = (float*)d_out;
    float* vqws = (float*)d_ws;                    // 8192 floats of scratch
    skill_fwd<<<dim3(B_), dim3(256), 0, stream>>>(p, out, vqws);
    vq_reduce<<<dim3(1), dim3(256), 0, stream>>>(vqws, out + VQ_OFF);
}
